// DenseCL_66039417143694
// MI455X (gfx1250) — matrix operand order repaired and checked by reading, not compiled, not run on hardware
//
#include <hip/hip_runtime.h>

// ---- problem constants -----------------------------------------------------
#define BSZ   8
#define DIMD  128
#define SS    1024
#define CF    512
#define KQ    16384
#define INV_TAU 5.0f
#define EPSN  1e-12f

// output layout (flat float32 elements; int targets are bitwise 0 == 0.0f)
#define OG_OFF  0                       // [8][16385]
#define TG_OFF  131080                  // [8]
#define OD_OFF  131088                  // [8][16385][1024]
#define TD_OFF  134357008               // [8][1024]

// ---- workspace layout (bytes) ---------------------------------------------
#define WS_FKT   0                      // bf16 [8][1024][512]   8 MB
#define WS_FQT   (8u*1024u*512u*2u)     // bf16 [8][1024][512]   8 MB
#define WS_MATCH (WS_FQT + 8u*1024u*512u*2u)          // int  [8][1024]
#define WS_DQGT  (WS_MATCH + 8u*1024u*4u)             // bf16 [8][1024][128]
#define WS_QDT   (WS_DQGT + 8u*1024u*128u*2u)         // bf16 [16384][128]
#define WS_GQN   (WS_QDT + 16384u*128u*2u)            // f32  [8][128]

typedef __attribute__((ext_vector_type(16))) __bf16 v16bf;
typedef __attribute__((ext_vector_type(8)))  __bf16 v8bf;
typedef __attribute__((ext_vector_type(8)))  float  v8f;

static __device__ __forceinline__ __bf16 f2bf(float f) {
  unsigned u = __builtin_bit_cast(unsigned, f);
  unsigned r = u + 0x7FFFu + ((u >> 16) & 1u);      // round-to-nearest-even
  return __builtin_bit_cast(__bf16, (unsigned short)(r >> 16));
}

// ---------------------------------------------------------------------------
// 1) g branch: normalize g_q/g_k per row (DIM=128), pos_g, stash g_qn.
//    One wave per batch row; 8 waves in one block.
__global__ void k_g(const float* __restrict__ g_q, const float* __restrict__ g_k,
                    float* __restrict__ out, float* __restrict__ g_qn) {
  int b    = threadIdx.x >> 5;
  int lane = threadIdx.x & 31;
  const float4 qa = *(const float4*)(g_q + b * DIMD + lane * 4);
  const float4 ka = *(const float4*)(g_k + b * DIMD + lane * 4);
  float sq = qa.x*qa.x + qa.y*qa.y + qa.z*qa.z + qa.w*qa.w;
  float sk = ka.x*ka.x + ka.y*ka.y + ka.z*ka.z + ka.w*ka.w;
  #pragma unroll
  for (int m = 16; m; m >>= 1) { sq += __shfl_xor(sq, m, 32); sk += __shfl_xor(sk, m, 32); }
  float iq = 1.0f / fmaxf(sqrtf(sq), EPSN);
  float ik = 1.0f / fmaxf(sqrtf(sk), EPSN);
  float4 qn; qn.x = qa.x*iq; qn.y = qa.y*iq; qn.z = qa.z*iq; qn.w = qa.w*iq;
  *(float4*)(g_qn + b * DIMD + lane * 4) = qn;
  float d = (qn.x*ka.x + qn.y*ka.y + qn.z*ka.z + qn.w*ka.w) * ik;
  #pragma unroll
  for (int m = 16; m; m >>= 1) d += __shfl_xor(d, m, 32);
  if (lane == 0) out[OG_OFF + (size_t)b * (KQ + 1)] = d * INV_TAU;
}

// ---------------------------------------------------------------------------
// 2) neg_g = g_qn @ queue_g  (tiny; coalesced over k)
__global__ void k_neg_g(const float* __restrict__ g_qn, const float* __restrict__ queue_g,
                        float* __restrict__ out) {
  int t = blockIdx.x * 256 + threadIdx.x;
  int b = t >> 14;          // /16384
  int k = t & (KQ - 1);
  const float* gq = g_qn + b * DIMD;
  float acc = 0.f;
  #pragma unroll 8
  for (int d = 0; d < DIMD; ++d) acc += gq[d] * queue_g[(size_t)d * KQ + k];
  out[OG_OFF + (size_t)b * (KQ + 1) + 1 + k] = acc * INV_TAU;
}

// ---------------------------------------------------------------------------
// 3) normalize feat over C and transpose [B][C][S] f32 -> [B][S][C] bf16
//    block = 256 thr handles a tile of 32 s positions; LDS transpose for
//    coalesced bf16 row writes.
__global__ void k_norm_feat(const float* __restrict__ feat, __bf16* __restrict__ fT) {
  __shared__ float  red[8][32];
  __shared__ __bf16 tile[32][CF];
  int b  = blockIdx.y;
  int s0 = blockIdx.x * 32;
  int sl = threadIdx.x & 31;
  int w  = threadIdx.x >> 5;
  const float* base = feat + ((size_t)b * CF) * SS + (s0 + sl);
  float ss = 0.f;
  for (int c = w; c < CF; c += 8) { float v = base[(size_t)c * SS]; ss += v * v; }
  red[w][sl] = ss;
  __syncthreads();
  float tot = 0.f;
  #pragma unroll
  for (int i = 0; i < 8; ++i) tot += red[i][sl];
  float scale = 1.0f / fmaxf(sqrtf(tot), EPSN);
  for (int c = w; c < CF; c += 8) tile[sl][c] = f2bf(base[(size_t)c * SS] * scale);
  __syncthreads();
  const unsigned* src = (const unsigned*)&tile[0][0];
  unsigned* dst = (unsigned*)(fT + ((size_t)b * SS + s0) * CF);
  for (int i = threadIdx.x; i < 32 * (CF / 2); i += 256) dst[i] = src[i];
}

// ---------------------------------------------------------------------------
// 4) fused cosine GEMM + argmax over key positions, per 16-wide j tile.
//    One wave per tile. Register budget: keep only the LOW half of the f_q
//    (B-matrix) fragments persistent (8 x v16bf = 64 VGPRs); the HIGH half is
//    re-read per i-tile through an opaque offset so LICM cannot hoist it back
//    into 128 persistent VGPRs (which previously spilled to scratch inside
//    the hot loop). The 16 KB f_q tile is WGP$-resident, so these are cheap
//    cache hits instead of scratch reloads.
__global__ void k_cos_argmax(const __bf16* __restrict__ fkT, const __bf16* __restrict__ fqT,
                             int* __restrict__ match) {
  int b    = blockIdx.y;
  int jt   = blockIdx.x * 4 + (threadIdx.x >> 5);
  int lane = threadIdx.x & 31;
  int n    = lane & 15;
  int koffB = (lane >> 4) * 16;                       // B: lanes 0-15 K=0..15, 16-31 K=16..31
  const __bf16* fq = fqT + ((size_t)b * SS + jt * 16 + n) * CF + koffB;
  v16bf Bf[8];
  #pragma unroll
  for (int cc = 0; cc < 8; ++cc) Bf[cc] = *(const v16bf*)(fq + cc * 32);

  int m     = lane & 15;
  int aoff  = (lane >> 4) * 16;                       // A: lanes 0-15 K=0..15, 16-31 K=16..31
  int mbase = (lane >> 4) << 3;                       // C frag: M = r (+8 for hi half)
  float bestv = -3.0e38f; int besti = 0;

  for (int it = 0; it < SS / 16; ++it) {
    const __bf16* fk = fkT + ((size_t)b * SS + it * 16 + m) * CF + aoff;
    int hoff = 8 * 32;                                // c = 256.. base offset
    asm volatile("" : "+v"(hoff));                    // opaque: block LICM hoist of hi-half B
    const __bf16* fqh = fq + hoff;
    v8f acc = {};
    #pragma unroll
    for (int cc = 0; cc < 8; ++cc) {                  // K = 0..255 : persistent B frags
      union { v16bf v; v8bf h[2]; } A;
      A.h[0] = *(const v8bf*)(fk + cc * 32);          // K = aoff .. aoff+7
      A.h[1] = *(const v8bf*)(fk + cc * 32 + 8);      // K = aoff+8 .. aoff+15
      acc = __builtin_amdgcn_wmma_f32_16x16x32_bf16(false, A.v, false, Bf[cc],
                                                    (short)0, acc, false, false);
    }
    #pragma unroll
    for (int cc = 0; cc < 8; ++cc) {                  // K = 256..511 : streamed B frags
      v16bf Bh = *(const v16bf*)(fqh + cc * 32);
      union { v16bf v; v8bf h[2]; } A;
      A.h[0] = *(const v8bf*)(fk + (8 + cc) * 32);
      A.h[1] = *(const v8bf*)(fk + (8 + cc) * 32 + 8);
      acc = __builtin_amdgcn_wmma_f32_16x16x32_bf16(false, A.v, false, Bh,
                                                    (short)0, acc, false, false);
    }
    #pragma unroll
    for (int r = 0; r < 8; ++r) {
      float v = acc[r];
      int   i = it * 16 + mbase + r;
      if (v > bestv) { bestv = v; besti = i; }
    }
  }
  float ov = __shfl_xor(bestv, 16, 32);
  int   oi = __shfl_xor(besti, 16, 32);
  if (ov > bestv || (ov == bestv && oi < besti)) { bestv = ov; besti = oi; }
  if (lane < 16) match[b * SS + jt * 16 + lane] = besti;
}

// ---------------------------------------------------------------------------
// 5) gather d_qn at match_idx -> bf16 [B][S][128]; pos_d -> output_d[:,0,:].
//    One wave per (b, z).
__global__ void k_gather(const float* __restrict__ d_q, const float* __restrict__ d_k,
                         const int* __restrict__ match, __bf16* __restrict__ dqgT,
                         float* __restrict__ out) {
  int b    = blockIdx.y;
  int z    = blockIdx.x * 8 + (threadIdx.x >> 5);
  int lane = threadIdx.x & 31;
  int idx  = match[b * SS + z];
  const float* dq = d_q + ((size_t)b * DIMD) * SS + idx;
  const float* dk = d_k + ((size_t)b * DIMD) * SS + z;
  float q[4], k[4], sq = 0.f, sk = 0.f;
  #pragma unroll
  for (int i = 0; i < 4; ++i) {
    q[i] = dq[(size_t)(lane + 32 * i) * SS];
    k[i] = dk[(size_t)(lane + 32 * i) * SS];
    sq += q[i] * q[i]; sk += k[i] * k[i];
  }
  #pragma unroll
  for (int m = 16; m; m >>= 1) { sq += __shfl_xor(sq, m, 32); sk += __shfl_xor(sk, m, 32); }
  float iq = 1.0f / fmaxf(sqrtf(sq), EPSN);
  float ik = 1.0f / fmaxf(sqrtf(sk), EPSN);
  float d = 0.f;
  #pragma unroll
  for (int i = 0; i < 4; ++i) d += q[i] * k[i];
  #pragma unroll
  for (int m = 16; m; m >>= 1) d += __shfl_xor(d, m, 32);
  __bf16* o = dqgT + ((size_t)b * SS + z) * DIMD;
  #pragma unroll
  for (int i = 0; i < 4; ++i) o[lane + 32 * i] = f2bf(q[i] * iq);
  if (lane == 0)
    out[OD_OFF + ((size_t)b * (KQ + 1)) * SS + z] = d * iq * ik * INV_TAU;
}

// ---------------------------------------------------------------------------
// 6) transpose queue_d [128][16384] f32 -> bf16 [16384][128] via LDS
__global__ void k_tq(const float* __restrict__ queue_d, __bf16* __restrict__ qdT) {
  __shared__ float t[32][33];
  int q0 = blockIdx.x * 32, d0 = blockIdx.y * 32;
  int lx = threadIdx.x & 31, ly = threadIdx.x >> 5;
  for (int r = ly; r < 32; r += 8)
    t[r][lx] = queue_d[(size_t)(d0 + r) * KQ + q0 + lx];
  __syncthreads();
  for (int r = ly; r < 32; r += 8)
    qdT[(size_t)(q0 + r) * DIMD + d0 + lx] = f2bf(t[lx][r]);
}

// ---------------------------------------------------------------------------
// 7) neg_d GEMM: [16384 x 1024], K=128 per batch, bf16 WMMA, f32 accumulate.
//    Wave tile = 64(M=q) x 16(N=z); B fragment reused across 4 M-subtiles.
//    Memory-bound on the 537 MB output_d store -> 16 WMMAs/wave is plenty.
__global__ void k_negd(const __bf16* __restrict__ qdT, const __bf16* __restrict__ dqgT,
                       float* __restrict__ out) {
  int b      = blockIdx.z;
  int z_base = blockIdx.x * 16;
  int w      = threadIdx.x >> 5;
  int lane   = threadIdx.x & 31;
  int q_base = blockIdx.y * 512 + w * 64;
  int n      = lane & 15;
  int m      = lane & 15;
  int koffB  = (lane >> 4) * 16;
  int koffA  = (lane >> 4) * 16;
  const __bf16* Bp = dqgT + ((size_t)b * SS + z_base + n) * DIMD + koffB;
  const __bf16* Ap = qdT + (size_t)(q_base + m) * DIMD + koffA;
  v8f acc[4] = {};
  #pragma unroll
  for (int kc = 0; kc < 4; ++kc) {
    v16bf Bf = *(const v16bf*)(Bp + kc * 32);
    #pragma unroll
    for (int mt = 0; mt < 4; ++mt) {
      const __bf16* a = Ap + (size_t)mt * 16 * DIMD + kc * 32;
      union { v16bf v; v8bf h[2]; } A;
      A.h[0] = *(const v8bf*)(a);
      A.h[1] = *(const v8bf*)(a + 8);
      acc[mt] = __builtin_amdgcn_wmma_f32_16x16x32_bf16(false, A.v, false, Bf,
                                                        (short)0, acc[mt], false, false);
    }
  }
  int mbase = (lane >> 4) << 3;
  float* o = out + OD_OFF + ((size_t)b * (KQ + 1) + 1 + q_base) * SS + z_base + n;
  #pragma unroll
  for (int mt = 0; mt < 4; ++mt)
    #pragma unroll
    for (int r = 0; r < 8; ++r)
      o[(size_t)(mt * 16 + mbase + r) * SS] = acc[mt][r] * INV_TAU;
}

// ---------------------------------------------------------------------------
// 8) int32 targets are all zero == 0.0f bit pattern
__global__ void k_zero(float* __restrict__ out) {
  int t = blockIdx.x * 256 + threadIdx.x;
  if (t < 8)    out[TG_OFF + t] = 0.0f;
  if (t < 8192) out[TD_OFF + t] = 0.0f;
}

// ---------------------------------------------------------------------------
extern "C" void kernel_launch(void* const* d_in, const int* in_sizes, int n_in,
                              void* d_out, int out_size, void* d_ws, size_t ws_size,
                              hipStream_t stream) {
  const float* g_q     = (const float*)d_in[0];
  const float* g_k     = (const float*)d_in[1];
  const float* d_q     = (const float*)d_in[2];
  const float* d_k     = (const float*)d_in[3];
  const float* feat_q  = (const float*)d_in[4];
  const float* feat_k  = (const float*)d_in[5];
  const float* queue_g = (const float*)d_in[6];
  const float* queue_d = (const float*)d_in[7];
  float* out = (float*)d_out;

  char* ws = (char*)d_ws;
  __bf16* fkT   = (__bf16*)(ws + WS_FKT);
  __bf16* fqT   = (__bf16*)(ws + WS_FQT);
  int*    match = (int*)   (ws + WS_MATCH);
  __bf16* dqgT  = (__bf16*)(ws + WS_DQGT);
  __bf16* qdT   = (__bf16*)(ws + WS_QDT);
  float*  g_qn  = (float*) (ws + WS_GQN);

  k_g        <<<1, 256, 0, stream>>>(g_q, g_k, out, g_qn);
  k_neg_g    <<<(BSZ * KQ) / 256, 256, 0, stream>>>(g_qn, queue_g, out);
  k_norm_feat<<<dim3(SS / 32, BSZ), 256, 0, stream>>>(feat_k, fkT);
  k_norm_feat<<<dim3(SS / 32, BSZ), 256, 0, stream>>>(feat_q, fqT);
  k_cos_argmax<<<dim3(16, BSZ), 128, 0, stream>>>(fkT, fqT, match);
  k_gather   <<<dim3(SS / 8, BSZ), 256, 0, stream>>>(d_q, d_k, match, dqgT, out);
  k_tq       <<<dim3(KQ / 32, DIMD / 32), 256, 0, stream>>>(queue_d, qdT);
  k_negd     <<<dim3(SS / 16, KQ / 512, BSZ), 256, 0, stream>>>(qdT, dqgT, out);
  k_zero     <<<8192 / 256, 256, 0, stream>>>(out);
}
